// ParallelMultiScaleROIExtractor_4818953306511
// MI455X (gfx1250) — compile-verified, matrix operand order checked by reading, not run on hardware
//
#include <hip/hip_runtime.h>
#include <hip/hip_bf16.h>

typedef __attribute__((ext_vector_type(16))) __bf16 v16bf;
typedef __attribute__((ext_vector_type(8)))  float  v8f;

union FragAB {
    v16bf v;
    uint4 q[2];
};

#define NBOX     8192
#define CCH      64
#define HH       120
#define WW       120
#define HIDDEN   128
#define OUTCOLS  64

__device__ __forceinline__ float bflo(unsigned int u) {
    union { unsigned int i; float f; } x; x.i = u << 16; return x.f;
}
__device__ __forceinline__ float bfhi(unsigned int u) {
    union { unsigned int i; float f; } x; x.i = u & 0xffff0000u; return x.f;
}

// ---------------------------------------------------------------------------
// Pre-pass: transpose feature map [C,H,W] f32 -> [H*W, C] bf16
// ---------------------------------------------------------------------------
__global__ __launch_bounds__(256)
void fm_transpose_kernel(const float* __restrict__ fm, __hip_bfloat16* __restrict__ fmhwc)
{
    int pix = blockIdx.x * blockDim.x + threadIdx.x;
    if (pix >= HH * WW) return;
#pragma unroll 8
    for (int c = 0; c < CCH; ++c)
        fmhwc[pix * CCH + c] = __float2bfloat16(fm[c * (HH * WW) + pix]);
}

// ---------------------------------------------------------------------------
// Pre-pass: per-channel mean of fm -> gf[64]
// ---------------------------------------------------------------------------
__global__ __launch_bounds__(256)
void fm_mean_kernel(const float* __restrict__ fm, float* __restrict__ gf)
{
    int c = blockIdx.x;
    float s = 0.f;
    for (int i = threadIdx.x; i < HH * WW; i += 256)
        s += fm[c * (HH * WW) + i];
    __shared__ float red[256];
    red[threadIdx.x] = s;
    __syncthreads();
    for (int off = 128; off > 0; off >>= 1) {
        if (threadIdx.x < off) red[threadIdx.x] += red[threadIdx.x + off];
        __syncthreads();
    }
    if (threadIdx.x == 0) gf[c] = red[0] * (1.0f / (float)(HH * WW));
}

// ---------------------------------------------------------------------------
// Pre-pass: f32 -> bf16 conversion (plain, for W2)
// ---------------------------------------------------------------------------
__global__ __launch_bounds__(256)
void cvt_bf16_kernel(const float* __restrict__ src, __hip_bfloat16* __restrict__ dst, int n)
{
    int i = blockIdx.x * blockDim.x + threadIdx.x;
    if (i < n) dst[i] = __float2bfloat16(src[i]);
}

// ---------------------------------------------------------------------------
// Pre-pass: W1 [128, 64*RR] f32, K-order c*RR+p  ->  bf16, K-order p*64+c
// ---------------------------------------------------------------------------
template<int RR>
__global__ __launch_bounds__(256)
void cvt_w1_perm_kernel(const float* __restrict__ src, __hip_bfloat16* __restrict__ dst)
{
    constexpr int FIN = CCH * RR;
    int i = blockIdx.x * blockDim.x + threadIdx.x;
    if (i >= HIDDEN * FIN) return;
    int n  = i / FIN;
    int kp = i - n * FIN;
    int p  = kp >> 6;          // k' = p*64 + c
    int c  = kp & 63;
    dst[i] = __float2bfloat16(src[n * FIN + c * RR + p]);
}

// ---------------------------------------------------------------------------
// g columns: out[:, 48+j] = bg[j] + sum_c gf[c]*Wg[j,c]  (broadcast over rows)
// ---------------------------------------------------------------------------
__global__ __launch_bounds__(256)
void g_write_kernel(const float* __restrict__ gf, const float* __restrict__ Wg,
                    const float* __restrict__ bg, float* __restrict__ out)
{
    int t = blockIdx.x * blockDim.x + threadIdx.x;   // NBOX*16 threads
    if (t >= NBOX * 16) return;
    int j = t & 15, row = t >> 4;
    float a = bg[j];
#pragma unroll 8
    for (int c = 0; c < CCH; ++c) a += gf[c] * Wg[j * CCH + c];
    out[(size_t)row * OUTCOLS + 48 + j] = a;
}

// ---------------------------------------------------------------------------
// Fused ROI-sample + MLP head.
//   One block = 32 boxes (two 16-row M-subtiles -> B fragment reused twice).
//   8 waves; GEMM1: wave w owns hidden cols [16w,16w+16) for both M-subtiles.
//   K is pixel-major: chunk q -> pixel p=q/2, channels (q&1)*32 .. +32.
//   A tile double-buffered; fill overlaps WMMA; one barrier per K-step.
// ---------------------------------------------------------------------------
template<int R, int OD, int COL0, int NW, int NT, int HID>
__global__ __launch_bounds__(256)
void head_kernel(const __hip_bfloat16* __restrict__ fmhwc,
                 const float*  __restrict__ boxes,
                 const __hip_bfloat16* __restrict__ W1,   // [128, FIN] bf16, K pixel-major
                 const float*  __restrict__ b1,           // [128] f32
                 const __hip_bfloat16* __restrict__ W2,   // [OD, 128] bf16
                 const float*  __restrict__ b2,           // [OD] f32
                 const float*  __restrict__ scale_w,
                 float* __restrict__ out)
{
    constexpr int RR   = R * R;
    constexpr int FIN  = CCH * RR;
    constexpr int NQ   = FIN / 32;   // = 2*RR K-chunks
    constexpr int APAD = 48;         // A row stride (elems): 96B, 16B aligned
    constexpr int HPAD = 136;        // h row stride (elems): 272B, 16B aligned
    constexpr int MT   = 32;         // boxes per block

    __shared__ __align__(16) float          sBox[MT][4];
    __shared__ __align__(16) __hip_bfloat16 sA[2][MT * APAD];
    __shared__ __align__(16) __hip_bfloat16 sH[MT * HPAD];

    const int tid   = threadIdx.x;
    const int lane  = tid & 31;
    const int wave  = tid >> 5;
    const int khalf = lane >> 4;          // 0: K 0-7/16-23, 1: K 8-15/24-31
    const int boxBase = blockIdx.x * MT;

    if (tid < MT) {
        const float SC = 119.0f / 960.0f;             // (W-1)/IMAGE_SIZE
        float x1 = boxes[(boxBase + tid) * 4 + 0];
        float y1 = boxes[(boxBase + tid) * 4 + 1];
        float x2 = boxes[(boxBase + tid) * 4 + 2];
        float y2 = boxes[(boxBase + tid) * 4 + 3];
        sBox[tid][0] = x1 * SC;                       // ix = ax + tx*dx
        sBox[tid][1] = (x2 - x1) * SC;
        sBox[tid][2] = y1 * SC;
        sBox[tid][3] = (y2 - y1) * SC;
    }
    __syncthreads();

    const float tstep = 1.0f / (float)(R - 1);

    // Fill one 32x32 bf16 A-tile: chunk q -> pixel p=q>>1, channel half q&1.
    // Each thread: rows (tid>>4) and (tid>>4)+16, two consecutive channels,
    // one 4B gather per corner per row (validity folded into weights).
    auto fillA = [&](int bsel, int qn) {
        int p   = qn >> 1;                 // uniform per chunk
        int py  = p / R;                   // scalar div by constant
        int px  = p - py * R;
        int row0  = tid >> 4;              // 0..15
        int chmem = ((qn & 1) << 4) + (tid & 15);  // dword index in pixel's 64ch
        const unsigned int* fmu = (const unsigned int*)fmhwc;
        unsigned int* dst = (unsigned int*)&sA[bsel][0];
#pragma unroll
        for (int sub = 0; sub < 2; ++sub) {
            int row = row0 + sub * 16;
            float ix = sBox[row][0] + (float)px * tstep * sBox[row][1];
            float iy = sBox[row][2] + (float)py * tstep * sBox[row][3];
            float x0f = floorf(ix), y0f = floorf(iy);
            float wx = ix - x0f, wy = iy - y0f;
            int x0 = (int)x0f,  y0 = (int)y0f;
            int x1i = x0 + 1,   y1i = y0 + 1;
            float mx0 = (x0  >= 0 && x0  < WW) ? 1.f : 0.f;
            float mx1 = (x1i >= 0 && x1i < WW) ? 1.f : 0.f;
            float my0 = (y0  >= 0 && y0  < HH) ? 1.f : 0.f;
            float my1 = (y1i >= 0 && y1i < HH) ? 1.f : 0.f;
            int x0c = min(max(x0, 0), WW - 1), x1c = min(max(x1i, 0), WW - 1);
            int y0c = min(max(y0, 0), HH - 1), y1c = min(max(y1i, 0), HH - 1);
            float w00 = (1.f - wx) * (1.f - wy) * mx0 * my0;
            float w01 = wx * (1.f - wy) * mx1 * my0;
            float w10 = (1.f - wx) * wy * mx0 * my1;
            float w11 = wx * wy * mx1 * my1;
            unsigned int u00 = fmu[(y0c * WW + x0c) * (CCH / 2) + chmem];
            unsigned int u01 = fmu[(y0c * WW + x1c) * (CCH / 2) + chmem];
            unsigned int u10 = fmu[(y1c * WW + x0c) * (CCH / 2) + chmem];
            unsigned int u11 = fmu[(y1c * WW + x1c) * (CCH / 2) + chmem];
            float va = w00 * bflo(u00) + w01 * bflo(u01) + w10 * bflo(u10) + w11 * bflo(u11);
            float vb = w00 * bfhi(u00) + w01 * bfhi(u01) + w10 * bfhi(u10) + w11 * bfhi(u11);
            union { unsigned int u; __hip_bfloat16 h[2]; } pk;
            pk.h[0] = __float2bfloat16(va);
            pk.h[1] = __float2bfloat16(vb);
            dst[row * (APAD / 2) + (tid & 15)] = pk.u;
        }
    };

    // ----------------- GEMM1: h = relu(x @ W1p^T + b1) -----------------
    v8f acc0 = {0.f, 0.f, 0.f, 0.f, 0.f, 0.f, 0.f, 0.f};
    v8f acc1 = {0.f, 0.f, 0.f, 0.f, 0.f, 0.f, 0.f, 0.f};
    const int n1 = wave * 16 + (lane & 15);       // hidden column for this lane

    fillA(0, 0);
    __syncthreads();

    for (int q = 0; q < NQ; ++q) {
        FragAB fa0, fa1, fb;
        {
            int row = lane & 15;
            const __hip_bfloat16* ab = sA[q & 1];
            fa0.q[0] = *(const uint4*)(&ab[ row       * APAD +      khalf * 8]);
            fa0.q[1] = *(const uint4*)(&ab[ row       * APAD + 16 + khalf * 8]);
            fa1.q[0] = *(const uint4*)(&ab[(row + 16) * APAD +      khalf * 8]);
            fa1.q[1] = *(const uint4*)(&ab[(row + 16) * APAD + 16 + khalf * 8]);
        }
        {
            const __hip_bfloat16* pb = W1 + (size_t)n1 * FIN + q * 32 + khalf * 8;
            fb.q[0] = *(const uint4*)(pb);
            fb.q[1] = *(const uint4*)(pb + 16);
        }
        acc0 = __builtin_amdgcn_wmma_f32_16x16x32_bf16(
                   false, fa0.v, false, fb.v, (short)0, acc0, false, false);
        acc1 = __builtin_amdgcn_wmma_f32_16x16x32_bf16(
                   false, fa1.v, false, fb.v, (short)0, acc1, false, false);
        if (q + 1 < NQ) fillA((q + 1) & 1, q + 1);   // overlaps WMMA execution
        __syncthreads();
    }

    // bias + relu, h -> LDS (bf16), both M-subtiles
    {
        float bias = b1[n1];
#pragma unroll
        for (int j = 0; j < 8; ++j) {
            int m = j + 8 * khalf;
            float v0 = acc0[j] + bias;
            float v1 = acc1[j] + bias;
            v0 = v0 > 0.f ? v0 : 0.f;
            v1 = v1 > 0.f ? v1 : 0.f;
            sH[ m       * HPAD + n1] = __float2bfloat16(v0);
            sH[(m + 16) * HPAD + n1] = __float2bfloat16(v1);
        }
    }
    __syncthreads();

    // ----------------- GEMM2: o = relu(h @ W2^T + b2) * scale -----------------
    if (wave < 2 * NT) {
        const int mt   = wave & 1;             // M-subtile
        const int nt   = wave >> 1;            // N-tile
        const int ncol = nt * 16 + (lane & 15);
        v8f acc2 = {0.f, 0.f, 0.f, 0.f, 0.f, 0.f, 0.f, 0.f};
#pragma unroll
        for (int k0 = 0; k0 < HIDDEN; k0 += 32) {
            FragAB fa, fb;
            int row = mt * 16 + (lane & 15);
            fa.q[0] = *(const uint4*)(&sH[row * HPAD + k0 +      khalf * 8]);
            fa.q[1] = *(const uint4*)(&sH[row * HPAD + k0 + 16 + khalf * 8]);
            if (ncol < OD) {
                const __hip_bfloat16* pb = W2 + ncol * HIDDEN + k0 + khalf * 8;
                fb.q[0] = *(const uint4*)(pb);
                fb.q[1] = *(const uint4*)(pb + 16);
            } else {
                fb.q[0] = make_uint4(0u, 0u, 0u, 0u);
                fb.q[1] = make_uint4(0u, 0u, 0u, 0u);
            }
            acc2 = __builtin_amdgcn_wmma_f32_16x16x32_bf16(
                       false, fa.v, false, fb.v, (short)0, acc2, false, false);
        }
        if (ncol < NW) {
            float bias = b2[ncol];
            float s    = scale_w[HID];
#pragma unroll
            for (int j = 0; j < 8; ++j) {
                int m = j + 8 * khalf;
                float v = acc2[j] + bias;
                v = v > 0.f ? v : 0.f;
                out[(size_t)(boxBase + mt * 16 + m) * OUTCOLS + COL0 + ncol] = v * s;
            }
        }
    }
}

// ---------------------------------------------------------------------------
// Host launcher
// ---------------------------------------------------------------------------
extern "C" void kernel_launch(void* const* d_in, const int* in_sizes, int n_in,
                              void* d_out, int out_size, void* d_ws, size_t ws_size,
                              hipStream_t stream)
{
    (void)in_sizes; (void)n_in; (void)out_size; (void)ws_size;

    const float* fm    = (const float*)d_in[0];
    const float* boxes = (const float*)d_in[1];
    const float* W1a = (const float*)d_in[2];  const float* b1a = (const float*)d_in[3];
    const float* W2a = (const float*)d_in[4];  const float* b2a = (const float*)d_in[5];
    const float* W1b = (const float*)d_in[6];  const float* b1b = (const float*)d_in[7];
    const float* W2b = (const float*)d_in[8];  const float* b2b = (const float*)d_in[9];
    const float* W1c = (const float*)d_in[10]; const float* b1c = (const float*)d_in[11];
    const float* W2c = (const float*)d_in[12]; const float* b2c = (const float*)d_in[13];
    const float* scale_w = (const float*)d_in[14];
    const float* Wg  = (const float*)d_in[15]; const float* bg  = (const float*)d_in[16];
    float* out = (float*)d_out;

    // workspace layout (256B aligned offsets)
    char* ws = (char*)d_ws;
    size_t off = 0;
    auto take = [&](size_t bytes) { void* p = ws + off; off = (off + bytes + 255) & ~(size_t)255; return p; };
    __hip_bfloat16* fmhwc = (__hip_bfloat16*)take((size_t)HH * WW * CCH * 2);
    float*          gf    = (float*)         take(CCH * 4);
    __hip_bfloat16* w1a   = (__hip_bfloat16*)take((size_t)HIDDEN * CCH * 9   * 2);
    __hip_bfloat16* w1b   = (__hip_bfloat16*)take((size_t)HIDDEN * CCH * 49  * 2);
    __hip_bfloat16* w1c   = (__hip_bfloat16*)take((size_t)HIDDEN * CCH * 121 * 2);
    __hip_bfloat16* w2a   = (__hip_bfloat16*)take((size_t)22 * HIDDEN * 2);
    __hip_bfloat16* w2b   = (__hip_bfloat16*)take((size_t)21 * HIDDEN * 2);
    __hip_bfloat16* w2c   = (__hip_bfloat16*)take((size_t)21 * HIDDEN * 2);

    // pre-passes
    fm_transpose_kernel<<<(HH * WW + 255) / 256, 256, 0, stream>>>(fm, fmhwc);
    fm_mean_kernel<<<CCH, 256, 0, stream>>>(fm, gf);

    cvt_w1_perm_kernel<9>  <<<(HIDDEN * CCH * 9   + 255) / 256, 256, 0, stream>>>(W1a, w1a);
    cvt_w1_perm_kernel<49> <<<(HIDDEN * CCH * 49  + 255) / 256, 256, 0, stream>>>(W1b, w1b);
    cvt_w1_perm_kernel<121><<<(HIDDEN * CCH * 121 + 255) / 256, 256, 0, stream>>>(W1c, w1c);
    cvt_bf16_kernel<<<(22 * HIDDEN + 255) / 256, 256, 0, stream>>>(W2a, w2a, 22 * HIDDEN);
    cvt_bf16_kernel<<<(21 * HIDDEN + 255) / 256, 256, 0, stream>>>(W2b, w2b, 21 * HIDDEN);
    cvt_bf16_kernel<<<(21 * HIDDEN + 255) / 256, 256, 0, stream>>>(W2c, w2c, 21 * HIDDEN);

    // fused heads: out cols [0,22) [22,43) [43,48)  (head c: only 5 cols survive)
    const int blocks = NBOX / 32;
    head_kernel< 3, 22,  0, 22, 2, 0><<<blocks, 256, 0, stream>>>(fmhwc, boxes, w1a, b1a, w2a, b2a, scale_w, out);
    head_kernel< 7, 21, 22, 21, 2, 1><<<blocks, 256, 0, stream>>>(fmhwc, boxes, w1b, b1b, w2b, b2b, scale_w, out);
    head_kernel<11, 21, 43,  5, 1, 2><<<blocks, 256, 0, stream>>>(fmhwc, boxes, w1c, b1c, w2c, b2c, scale_w, out);

    // g broadcast: out cols [48,64)
    g_write_kernel<<<(NBOX * 16 + 255) / 256, 256, 0, stream>>>(gf, Wg, bg, out);
}
